// TripartiteAttention_3848290697328
// MI455X (gfx1250) — compile-verified, matrix operand order checked by reading, not run on hardware
//
#include <hip/hip_runtime.h>
#include <hip/hip_bf16.h>
#include <math.h>

// MI455X / gfx1250, wave32, WMMA 16x16x32 f16->f32 path.
typedef __attribute__((ext_vector_type(16))) _Float16 v16h;
typedef __attribute__((ext_vector_type(8)))  _Float16 v8h;
typedef __attribute__((ext_vector_type(8)))  float    v8f;

#define SQRT_SCALE 0.42044820762685725f  // (1/sqrt(32))^(1/2); folded into Q,Kb,Kc

// --- WMMA fragment helpers ------------------------------------------------
// 16-bit A-matrix 16x32 layout (ISA 7.12.2): lane r (r<16) = row r, K halves
// 0..7 then 16..23; lanes 16..31 = same rows, K halves 8..15 then 24..31.
// B (K x N) per-lane layout is the transpose-symmetric pattern, so loading B
// from a row-major N x K array with the SAME loader gives A * B^T semantics.
__device__ __forceinline__ v16h load_frag16(const _Float16* __restrict__ src,
                                            int ld, int row0, int k0, int lane) {
  const int r = lane & 15, hi = lane >> 4;
  const _Float16* p = src + (size_t)(row0 + r) * ld + k0 + hi * 8;
  v8h lo = *(const v8h*)(p);
  v8h hh = *(const v8h*)(p + 16);
  v16h a;
#pragma unroll
  for (int i = 0; i < 8; ++i) { a[i] = lo[i]; a[i + 8] = hh[i]; }
  return a;
}

__device__ __forceinline__ v8f wmma_tile_f16(const _Float16* __restrict__ A, int lda,
                                             const _Float16* __restrict__ B, int ldb,
                                             int row0, int col0, int K, int lane) {
  v8f acc;
#pragma unroll
  for (int i = 0; i < 8; ++i) acc[i] = 0.0f;
  for (int k = 0; k < K; k += 32) {
    v16h a = load_frag16(A, lda, row0, k, lane);
    v16h b = load_frag16(B, ldb, col0, k, lane);
    // (neg_a, A, neg_b, B, c_mod, C, reuse_a, reuse_b)
    acc = __builtin_amdgcn_wmma_f32_16x16x32_f16(false, a, false, b,
                                                 (short)0, acc, false, false);
  }
  return acc;
}

// C/D f32 16x16 layout: lane l: col = l&15, rows = vgpr + (l>>4)*8.
__device__ __forceinline__ void store_tile_f32(float* __restrict__ dst, int ldo,
                                               int row0, int col0, int lane, v8f d) {
  const int r = lane & 15, hi = lane >> 4;
#pragma unroll
  for (int v = 0; v < 8; ++v)
    dst[(size_t)(row0 + hi * 8 + v) * ldo + col0 + r] = d[v];
}

__device__ __forceinline__ void store_tile_f16s(_Float16* __restrict__ dst, int ldo,
                                                int row0, int col0, int lane, v8f d, float s) {
  const int r = lane & 15, hi = lane >> 4;
#pragma unroll
  for (int v = 0; v < 8; ++v)
    dst[(size_t)(row0 + hi * 8 + v) * ldo + col0 + r] = (_Float16)(d[v] * s);
}

// --- Stage 1: fp32 -> fp16 conversions (W matrices transposed to N x K) ----
__global__ void convert_kernel(const float* __restrict__ Ha, const float* __restrict__ Hb,
                               const float* __restrict__ Hc, const float* __restrict__ Wq,
                               const float* __restrict__ Wk, const float* __restrict__ Wv,
                               _Float16* Ha16, _Float16* Hb16, _Float16* Hc16,
                               _Float16* Wq16t, _Float16* Wk16t, _Float16* Wv16t) {
  int i = blockIdx.x * blockDim.x + threadIdx.x;
  if (i < 16384) { Ha16[i] = (_Float16)Ha[i]; return; }
  i -= 16384;
  if (i < 32768) { Hb16[i] = (_Float16)Hb[i]; return; }
  i -= 32768;
  if (i < 32768) { Hc16[i] = (_Float16)Hc[i]; return; }
  i -= 32768;
  if (i < 32768) { int n = i >> 8, k = i & 255; Wq16t[i] = (_Float16)Wq[k * 128 + n]; return; }
  i -= 32768;
  if (i < 32768) { int n = i >> 8, k = i & 255; Wk16t[i] = (_Float16)Wk[k * 128 + n]; return; }
  i -= 32768;
  if (i < 32768) { int n = i >> 8, k = i & 255; Wv16t[i] = (_Float16)Wv[k * 128 + n]; }
}

// --- Stage 2: 5 projection GEMMs (K=256) via WMMA; one wave per 16x16 tile -
__global__ void proj_kernel(const _Float16* Ha16, const _Float16* Hb16, const _Float16* Hc16,
                            const _Float16* Wq16t, const _Float16* Wk16t, const _Float16* Wv16t,
                            _Float16* Q16, _Float16* Kb16, _Float16* Kc16,
                            float* Vb, float* Vc) {
  const int wid = (blockIdx.x * blockDim.x + threadIdx.x) >> 5;
  const int lane = threadIdx.x & 31;
  if (wid >= 288) return;               // 32 + 4*64 tiles, uniform per wave
  int which, t;
  if (wid < 32)       { which = 0; t = wid; }
  else if (wid < 96)  { which = 1; t = wid - 32; }
  else if (wid < 160) { which = 2; t = wid - 96; }
  else if (wid < 224) { which = 3; t = wid - 160; }
  else                { which = 4; t = wid - 224; }
  const int mt = t >> 3, nt = t & 7;
  const _Float16 *A, *B;
  switch (which) {
    case 0:  A = Ha16; B = Wq16t; break;
    case 1:  A = Hb16; B = Wk16t; break;
    case 2:  A = Hc16; B = Wk16t; break;
    case 3:  A = Hb16; B = Wv16t; break;
    default: A = Hc16; B = Wv16t; break;
  }
  v8f acc = wmma_tile_f16(A, 256, B, 256, mt * 16, nt * 16, 256, lane);
  switch (which) {
    case 0:  store_tile_f16s(Q16,  128, mt * 16, nt * 16, lane, acc, SQRT_SCALE); break;
    case 1:  store_tile_f16s(Kb16, 128, mt * 16, nt * 16, lane, acc, SQRT_SCALE); break;
    case 2:  store_tile_f16s(Kc16, 128, mt * 16, nt * 16, lane, acc, SQRT_SCALE); break;
    case 3:  store_tile_f32(Vb, 128, mt * 16, nt * 16, lane, acc); break;
    default: store_tile_f32(Vc, 128, mt * 16, nt * 16, lane, acc); break;
  }
}

// --- Stage 3: per-head score GEMMs (K=32, single WMMA per tile) ------------
__global__ void score_kernel(const _Float16* Q16, const _Float16* Kb16, const _Float16* Kc16,
                             float* sab, float* sac, float* sbc) {
  const int wid = (blockIdx.x * blockDim.x + threadIdx.x) >> 5;
  const int lane = threadIdx.x & 31;
  if (wid >= 512) return;
  if (wid < 256) {                                   // s_ab / s_ac : 64x128 per head
    const int grp = wid >> 7;
    const int t = wid & 127;
    const int h = t >> 5, tt = t & 31;
    const int mt = tt >> 3, nt = tt & 7;
    const _Float16* A = Q16 + h * 32;
    const _Float16* B = (grp == 0 ? Kb16 : Kc16) + h * 32;
    v8f acc = wmma_tile_f16(A, 128, B, 128, mt * 16, nt * 16, 32, lane);
    float* out = (grp == 0 ? sab : sac) + h * (64 * 128);
    store_tile_f32(out, 128, mt * 16, nt * 16, lane, acc);
  } else {                                           // s_bc : 128x128 per head
    const int t = wid - 256;
    const int h = t >> 6, tt = t & 63;
    const int mt = tt >> 3, nt = tt & 7;
    v8f acc = wmma_tile_f16(Kb16 + h * 32, 128, Kc16 + h * 32, 128,
                            mt * 16, nt * 16, 32, lane);
    store_tile_f32(sbc + h * (128 * 128), 128, mt * 16, nt * 16, lane, acc);
  }
}

// --- Stage 4: stabilization maxima -----------------------------------------
__global__ void maxab_kernel(const float* sab, const float* sac, float* mab, float* mac) {
  const int tid = blockIdx.x * blockDim.x + threadIdx.x;
  if (tid < 256) {
    const int a = tid >> 2, h = tid & 3;
    const float* p = sab + (h * 64 + a) * 128;
    float m = -3.4e38f;
    for (int b = 0; b < 128; ++b) m = fmaxf(m, p[b]);
    mab[tid] = m;
  } else if (tid < 512) {
    const int t = tid - 256;
    const int a = t >> 2, h = t & 3;
    const float* p = sac + (h * 64 + a) * 128;
    float m = -3.4e38f;
    for (int c = 0; c < 128; ++c) m = fmaxf(m, p[c]);
    mac[t] = m;
  }
}

__global__ void maxbc_kernel(const float* sbc, float* mbc) {
  __shared__ float red[256];
  const int h = blockIdx.x;
  float m = -3.4e38f;
  for (int i = threadIdx.x; i < 16384; i += 256) m = fmaxf(m, sbc[h * 16384 + i]);
  red[threadIdx.x] = m;
  __syncthreads();
  for (int s = 128; s > 0; s >>= 1) {
    if (threadIdx.x < (unsigned)s) red[threadIdx.x] = fmaxf(red[threadIdx.x], red[threadIdx.x + s]);
    __syncthreads();
  }
  if (threadIdx.x == 0) mbc[h] = red[0];
}

// --- Stage 5: exponentials (f32 + f16 copies for the Z-GEMM) ---------------
__global__ void exp_kernel(const float* sab, const float* sac, const float* sbc,
                           const float* mab, const float* mac, const float* mbc,
                           float* eab, float* eac, _Float16* eac16,
                           float* Ebc, _Float16* Ebc16) {
  int i = blockIdx.x * blockDim.x + threadIdx.x;
  if (i < 32768) {
    const int h = i >> 13, a = (i >> 7) & 63;
    eab[i] = expf(sab[i] - mab[a * 4 + h]);
    return;
  }
  i -= 32768;
  if (i < 32768) {
    const int h = i >> 13, a = (i >> 7) & 63;
    const float e = expf(sac[i] - mac[a * 4 + h]);
    eac[i] = e; eac16[i] = (_Float16)e;
    return;
  }
  i -= 32768;
  if (i < 65536) {
    const int h = i >> 14;
    const float e = expf(sbc[i] - mbc[h]);
    Ebc[i] = e; Ebc16[i] = (_Float16)e;
  }
}

// --- Stage 6: fc_w partial dots: pb[h][b][j], pcx[h][c] (j = c&3) ----------
__global__ void pv_kernel(const float* Vb, const float* Vc, const float* fcw,
                          float* pb, float* pcx) {
  const int t = blockIdx.x * blockDim.x + threadIdx.x;
  if (t < 2048) {
    const int j = t & 3, b = (t >> 2) & 127, h = t >> 9;
    const float* v = Vb + b * 128 + h * 32;
    const float* w = fcw + j * 32;
    float s = 0.f;
#pragma unroll
    for (int d = 0; d < 32; ++d) s += v[d] * w[d];
    pb[t] = s;
  } else if (t < 2560) {
    const int u = t - 2048;
    const int c = u & 127, h = u >> 7, j = c & 3;
    const float* v = Vc + c * 128 + h * 32;
    const float* w = fcw + j * 32;
    float s = 0.f;
#pragma unroll
    for (int d = 0; d < 32; ++d) s += v[d] * w[d];
    pcx[u] = s;
  }
}

// --- Stage 7: partition-function GEMM U[a,b] = sum_c eac[a,c]*Ebc[b,c] -----
__global__ void u_kernel(const _Float16* eac16, const _Float16* Ebc16, float* U) {
  const int wid = (blockIdx.x * blockDim.x + threadIdx.x) >> 5;
  const int lane = threadIdx.x & 31;
  if (wid >= 128) return;
  const int h = wid >> 5, t = wid & 31, mt = t >> 3, nt = t & 7;
  v8f acc = wmma_tile_f16(eac16 + h * 8192, 128, Ebc16 + h * 16384, 128,
                          mt * 16, nt * 16, 128, lane);
  store_tile_f32(U + h * 8192, 128, mt * 16, nt * 16, lane, acc);
}

// --- Stage 8: Z[a,h] = sum_b eab * U -------------------------------------
__global__ void z_kernel(const float* eab, const float* U, float* Z) {
  const int tid = blockIdx.x * blockDim.x + threadIdx.x;
  if (tid >= 256) return;
  const int a = tid >> 2, h = tid & 3;
  const float* pe = eab + (h * 64 + a) * 128;
  const float* pu = U + (h * 64 + a) * 128;
  float z = 0.f;
  for (int b = 0; b < 128; ++b) z += pe[b] * pu[b];
  Z[tid] = z;
}

// --- Stage 9: fused output (reshape decomposition + sigmoid) ---------------
// out[a,b',c'] = sigmoid( (eab[a,h,b]/Z) * sum_j eac[a,h,cb+j]*Ebc[h,b,cb+j]
//                         *(pb[h,b,j]+pcx[h,cb+j]) + fc_b )
// with h=b'>>5, b=(b'&31)*4+(c'>>5), cb=(c'&31)*4.
__global__ void output_kernel(const float* eab, const float* eac, const float* Ebc,
                              const float* Z, const float* pb, const float* pcx,
                              const float* fcb, float* out) {
  const int o = blockIdx.x * blockDim.x + threadIdx.x;
  if (o >= 64 * 128 * 128) return;
  const int a = o >> 14;
  const int bp = (o >> 7) & 127;
  const int cp = o & 127;
  const int h = bp >> 5;
  const int b = ((bp & 31) << 2) | (cp >> 5);
  const int cb = (cp & 31) << 2;
  const float g = eab[(h * 64 + a) * 128 + b] / Z[a * 4 + h];
  const float4 ec = *(const float4*)(eac + (h * 64 + a) * 128 + cb);
  const float4 eb = *(const float4*)(Ebc + (h * 128 + b) * 128 + cb);
  const float4 vb = *(const float4*)(pb + (h * 128 + b) * 4);
  const float4 vc = *(const float4*)(pcx + h * 128 + cb);
  const float acc = ec.x * eb.x * (vb.x + vc.x)
                  + ec.y * eb.y * (vb.y + vc.y)
                  + ec.z * eb.z * (vb.z + vc.z)
                  + ec.w * eb.w * (vb.w + vc.w);
  const float x = g * acc + fcb[0];
  out[o] = 1.0f / (1.0f + expf(-x));
}

extern "C" void kernel_launch(void* const* d_in, const int* in_sizes, int n_in,
                              void* d_out, int out_size, void* d_ws, size_t ws_size,
                              hipStream_t stream) {
  (void)in_sizes; (void)n_in; (void)out_size; (void)ws_size;
  const float* Ha  = (const float*)d_in[0];
  const float* Hb  = (const float*)d_in[1];
  const float* Hc  = (const float*)d_in[2];
  const float* Wq  = (const float*)d_in[3];
  const float* Wk  = (const float*)d_in[4];
  const float* Wv  = (const float*)d_in[5];
  const float* fcw = (const float*)d_in[6];
  const float* fcb = (const float*)d_in[7];
  float* out = (float*)d_out;

  char* w = (char*)d_ws;
  auto take = [&](size_t bytes) -> char* {
    char* p = w;
    w += (bytes + 255) & ~(size_t)255;
    return p;
  };
  _Float16* Ha16  = (_Float16*)take(16384 * 2);
  _Float16* Hb16  = (_Float16*)take(32768 * 2);
  _Float16* Hc16  = (_Float16*)take(32768 * 2);
  _Float16* Wq16t = (_Float16*)take(32768 * 2);
  _Float16* Wk16t = (_Float16*)take(32768 * 2);
  _Float16* Wv16t = (_Float16*)take(32768 * 2);
  _Float16* Q16   = (_Float16*)take(8192 * 2);
  _Float16* Kb16  = (_Float16*)take(16384 * 2);
  _Float16* Kc16  = (_Float16*)take(16384 * 2);
  float* Vb   = (float*)take(16384 * 4);
  float* Vc   = (float*)take(16384 * 4);
  float* sab  = (float*)take(32768 * 4);
  float* sac  = (float*)take(32768 * 4);
  float* sbc  = (float*)take(65536 * 4);
  float* eab  = (float*)take(32768 * 4);
  float* eac  = (float*)take(32768 * 4);
  _Float16* eac16 = (_Float16*)take(32768 * 2);
  float* Ebc  = (float*)take(65536 * 4);
  _Float16* Ebc16 = (_Float16*)take(65536 * 2);
  float* U    = (float*)take(32768 * 4);
  float* mab  = (float*)take(256 * 4);
  float* mac  = (float*)take(256 * 4);
  float* mbc  = (float*)take(16);
  float* Z    = (float*)take(256 * 4);
  float* pb   = (float*)take(2048 * 4);
  float* pcx  = (float*)take(512 * 4);

  convert_kernel<<<704, 256, 0, stream>>>(Ha, Hb, Hc, Wq, Wk, Wv,
                                          Ha16, Hb16, Hc16, Wq16t, Wk16t, Wv16t);
  proj_kernel<<<36, 256, 0, stream>>>(Ha16, Hb16, Hc16, Wq16t, Wk16t, Wv16t,
                                      Q16, Kb16, Kc16, Vb, Vc);
  score_kernel<<<64, 256, 0, stream>>>(Q16, Kb16, Kc16, sab, sac, sbc);
  maxab_kernel<<<2, 256, 0, stream>>>(sab, sac, mab, mac);
  maxbc_kernel<<<4, 256, 0, stream>>>(sbc, mbc);
  exp_kernel<<<512, 256, 0, stream>>>(sab, sac, sbc, mab, mac, mbc,
                                      eab, eac, eac16, Ebc, Ebc16);
  pv_kernel<<<10, 256, 0, stream>>>(Vb, Vc, fcw, pb, pcx);
  u_kernel<<<16, 256, 0, stream>>>(eac16, Ebc16, U);
  z_kernel<<<1, 256, 0, stream>>>(eab, U, Z);
  output_kernel<<<4096, 256, 0, stream>>>(eab, eac, Ebc, Z, pb, pcx, fcb, out);
}
